// GLALOBClsToken_84748294684695
// MI455X (gfx1250) — matrix operand order repaired and checked by reading, not compiled, not run on hardware
//
#include <hip/hip_runtime.h>
#include <hip/hip_bf16.h>
#include <math.h>

#define Bz   32
#define Sz   512
#define Fz   40
#define Dz   512
#define Hz   8
#define Lz   4
#define Tz   513
#define DKz  256
#define DVz  512
#define HKz  32
#define HVz  64
#define GLRz 16
#define MLPz 2048
#define Mz   (Bz * Tz)   // 16416 = 32 * 513

typedef __attribute__((ext_vector_type(16))) _Float16 f16x16;
typedef __attribute__((ext_vector_type(8)))  float    f32x8;

// ---------------- transpose + cast fp32 (K,N) -> fp16 (N,K) ----------------
__global__ void tcast_f16(const float* __restrict__ src, _Float16* __restrict__ dst,
                          int K, int N) {
    int idx = blockIdx.x * 256 + threadIdx.x;
    if (idx >= K * N) return;
    int k = idx / N, n = idx - k * N;
    dst[(size_t)n * K + k] = (_Float16)src[idx];
}

// ---------------- BiN (dual z-norm, ddof=1, small-std -> 1) ----------------
__global__ void bin_norm_kernel(const float* __restrict__ x,
                                const float* __restrict__ l1, const float* __restrict__ b1,
                                const float* __restrict__ l2, const float* __restrict__ b2,
                                const float* __restrict__ y1p, const float* __restrict__ y2p,
                                float* __restrict__ xn) {
    int b = blockIdx.x, tid = threadIdx.x;           // 256 threads
    __shared__ float m2[Fz], is2[Fz];                // per-feature stats over time
    __shared__ float m1[Sz], is1[Sz];                // per-time stats over features
    const float* xb = x + (size_t)b * Sz * Fz;
    if (tid < Fz) {
        float s = 0.f, q = 0.f;
        for (int t = 0; t < Sz; ++t) { float v = xb[t * Fz + tid]; s += v; q += v * v; }
        float mean = s / Sz;
        float var  = fmaxf((q - Sz * mean * mean) / (Sz - 1), 0.f);
        float st   = sqrtf(var); if (st < 1e-4f) st = 1.f;
        m2[tid] = mean; is2[tid] = 1.f / st;
    }
    for (int t = tid; t < Sz; t += 256) {
        float s = 0.f, q = 0.f;
        for (int f = 0; f < Fz; ++f) { float v = xb[t * Fz + f]; s += v; q += v * v; }
        float mean = s / Fz;
        float var  = fmaxf((q - Fz * mean * mean) / (Fz - 1), 0.f);
        float st   = sqrtf(var); if (st < 1e-4f) st = 1.f;
        m1[t] = mean; is1[t] = 1.f / st;
    }
    __syncthreads();
    float y1 = *y1p, y2 = *y2p;
    for (int idx = tid; idx < Sz * Fz; idx += 256) {
        int t = idx / Fz, f = idx - t * Fz;
        float v  = xb[idx];
        float X1 = l1[t] * ((v - m1[t]) * is1[t]) + b1[t];
        float X2 = l2[f] * ((v - m2[f]) * is2[f]) + b2[f];
        xn[(size_t)b * Sz * Fz + idx] = y1 * X1 + y2 * X2;
    }
}

// ---------------- embed + cls token + sinusoidal positions ----------------
__global__ void embed_kernel(const float* __restrict__ xn,
                             const float* __restrict__ emb_w, const float* __restrict__ emb_b,
                             const float* __restrict__ cls, float* __restrict__ h) {
    int s = blockIdx.x, b = blockIdx.y, tid = threadIdx.x;   // 128 threads
    __shared__ float xr[Fz];
    if (s < Sz && tid < Fz) xr[tid] = xn[((size_t)b * Sz + s) * Fz + tid];
    __syncthreads();
    for (int d = tid; d < Dz; d += 128) {
        float acc;
        if (s < Sz) {
            acc = emb_b[d];
            for (int f = 0; f < Fz; ++f) acc += xr[f] * emb_w[f * Dz + d];
        } else {
            acc = cls[d];
        }
        int   i   = d >> 1;
        float den = __powf(10000.f, (2.f * i) / Dz);
        float ang = (float)s / den;
        acc += (d & 1) ? cosf(ang) : sinf(ang);
        h[((size_t)b * Tz + s) * Dz + d] = acc;
    }
}

// ---------------- LayerNorm (biased var, eps 1e-5) -> fp16 ----------------
__global__ void layernorm_kernel(const float* __restrict__ h,
                                 const float* __restrict__ w, const float* __restrict__ bw,
                                 _Float16* __restrict__ out) {
    int row = blockIdx.x, tid = threadIdx.x;   // 128 threads, D=512
    const float* hr = h + (size_t)row * Dz;
    float lv[4], s = 0.f, q = 0.f;
#pragma unroll
    for (int j = 0; j < 4; ++j) { lv[j] = hr[tid + j * 128]; s += lv[j]; q += lv[j] * lv[j]; }
    __shared__ float rs[128], rq[128];
    rs[tid] = s; rq[tid] = q; __syncthreads();
    for (int st = 64; st > 0; st >>= 1) {
        if (tid < st) { rs[tid] += rs[tid + st]; rq[tid] += rq[tid + st]; }
        __syncthreads();
    }
    float mean = rs[0] / Dz;
    float var  = fmaxf(rq[0] / Dz - mean * mean, 0.f);
    float inv  = rsqrtf(var + 1e-5f);
#pragma unroll
    for (int j = 0; j < 4; ++j) {
        int d = tid + j * 128;
        out[(size_t)row * Dz + d] = (_Float16)((lv[j] - mean) * inv * w[d] + bw[d]);
    }
}

// ------ WMMA GEMM. PAIR=true: each wave computes a 32x32 macro-tile (2x2
// 16x16 tiles, branch-free 4-WMMA K-loop). PAIR=false: two 16x16 tiles (32x16)/wave
// (only used for the tiny N=16 GEMM).
// C[M,N] = act(alpha * A[M,K] @ Wt[N,K]^T + bias) (+resid), out fp32 and/or fp16
template <bool PAIR>
__global__ void gemm_wmma_t(const _Float16* __restrict__ A,
                            const _Float16* __restrict__ Wt,
                            const float* __restrict__ bias,
                            const float* __restrict__ resid,
                            float* __restrict__ outF,
                            _Float16* __restrict__ outB,
                            int M, int N, int K, float alpha, int act) {
    int lane = threadIdx.x & 31;
    int wave = threadIdx.x >> 5;
    int tM0  = blockIdx.x * 2;                                   // M multiple of 32
    int tN0  = PAIR ? (blockIdx.y * 4 + wave) * 2 : (blockIdx.y * 4 + wave);
    if (tN0 * 16 >= N) return;
    int hi  = lane >> 4;     // lane group (K-phase)
    int l15 = lane & 15;
    // A frag: lane row m = l15; elems 0..7 <- K kb..kb+7, 8..15 <- K kb+16..kb+23, kb=hi*8
    const _Float16* Ar0 = A  + (size_t)(tM0 * 16 + l15) * K + hi * 8;
    const _Float16* Ar1 = Ar0 + (size_t)16 * K;
    // B frag: lane col n = l15; elems 0..7 <- K kb..kb+7, 8..15 <- K kb+16..kb+23, kb=hi*8
    const _Float16* Br0 = Wt + (size_t)(tN0 * 16 + l15) * K + hi * 8;
    const _Float16* Br1 = Br0 + (size_t)16 * K;                  // used iff PAIR
    f32x8 c00 = {}, c01 = {}, c10 = {}, c11 = {};
    for (int k0 = 0; k0 < K; k0 += 32) {
        f16x16 a0, a1, b0;
#pragma unroll
        for (int j = 0; j < 8; ++j) {
            a0[j] = Ar0[k0 + j]; a0[j + 8] = Ar0[k0 + 16 + j];
            a1[j] = Ar1[k0 + j]; a1[j + 8] = Ar1[k0 + 16 + j];
        }
#pragma unroll
        for (int j = 0; j < 8; ++j) { b0[j] = Br0[k0 + j]; b0[j + 8] = Br0[k0 + 16 + j]; }
        c00 = __builtin_amdgcn_wmma_f32_16x16x32_f16(false, a0, false, b0, (short)0, c00, false, false);
        c10 = __builtin_amdgcn_wmma_f32_16x16x32_f16(false, a1, false, b0, (short)0, c10, false, false);
        if constexpr (PAIR) {
            f16x16 b1;
#pragma unroll
            for (int j = 0; j < 8; ++j) { b1[j] = Br1[k0 + j]; b1[j + 8] = Br1[k0 + 16 + j]; }
            c01 = __builtin_amdgcn_wmma_f32_16x16x32_f16(false, a0, false, b1, (short)0, c01, false, false);
            c11 = __builtin_amdgcn_wmma_f32_16x16x32_f16(false, a1, false, b1, (short)0, c11, false, false);
        }
    }
    // C/D layout: lane = N col (l15), vgpr r = M row (+8 for lanes 16..31)
    auto epi = [&](int tM, int tN, const f32x8& c) {
        int   n  = tN * 16 + l15;
        int   mb = tM * 16 + hi * 8;
        float bv = bias ? bias[n] : 0.f;
#pragma unroll
        for (int r = 0; r < 8; ++r) {
            float v = alpha * c[r] + bv;
            if (act == 1) v = 0.5f * v * (1.f + erff(v * 0.70710678118654752f)); // exact GELU
            size_t off = (size_t)(mb + r) * N + n;
            if (resid) v += resid[off];
            if (outF) outF[off] = v;
            if (outB) outB[off] = (_Float16)v;
        }
    };
    epi(tM0,     tN0, c00);
    epi(tM0 + 1, tN0, c10);
    if constexpr (PAIR) {
        epi(tM0,     tN0 + 1, c01);
        epi(tM0 + 1, tN0 + 1, c11);
    }
}

// ---------------- gate stage 2: logsigmoid(glow@w2 + b2)/16 ----------------
__global__ void gk2_kernel(const float* __restrict__ glow,
                           const float* __restrict__ w2, const float* __restrict__ b2,
                           float* __restrict__ gk) {
    int m = blockIdx.x, n = threadIdx.x;   // 256 threads = DK
    __shared__ float gl[GLRz];
    if (n < GLRz) gl[n] = glow[(size_t)m * GLRz + n];
    __syncthreads();
    float z = b2[n];
#pragma unroll
    for (int r = 0; r < GLRz; ++r) z += gl[r] * w2[r * DKz + n];
    float ls = fminf(z, 0.f) - log1pf(expf(-fabsf(z)));   // stable log-sigmoid
    gk[(size_t)m * DKz + n] = ls * (1.f / 16.f);
}

// ---------------- GLA recurrence + head-RMSNorm + swish gate ----------------
__global__ void gla_scan_kernel(const float* __restrict__ q,  const float* __restrict__ k,
                                const float* __restrict__ v,  const float* __restrict__ g,
                                const float* __restrict__ gk, const float* __restrict__ gnw,
                                _Float16* __restrict__ obf) {
    int bh = blockIdx.x;            // b*H + h
    int b = bh >> 3, hh = bh & 7;
    int tid = threadIdx.x;          // 256 threads
    int vv = tid & 63, kg = tid >> 6;   // state slice: k = kg*8+j, col = vv
    __shared__ float qs[HKz], ks[HKz], es[HKz], vs[HVz], gs[HVz];
    __shared__ float op[4][HVz], red[HVz];
    float st[8];
#pragma unroll
    for (int j = 0; j < 8; ++j) st[j] = 0.f;
    for (int t = 0; t < Tz; ++t) {
        size_t bq = ((size_t)(b * Tz + t)) * DKz + hh * HKz;
        size_t bv = ((size_t)(b * Tz + t)) * DVz + hh * HVz;
        if (tid < 32)      { qs[tid] = q[bq + tid]; ks[tid] = k[bq + tid]; es[tid] = expf(gk[bq + tid]); }
        else if (tid < 96) { int j = tid - 32; vs[j] = v[bv + j]; gs[j] = g[bv + j]; }
        __syncthreads();
        float acc = 0.f, vval = vs[vv];
#pragma unroll
        for (int j = 0; j < 8; ++j) {
            int kk = kg * 8 + j;
            st[j] = st[j] * es[kk] + ks[kk] * vval;
            acc  += qs[kk] * st[j];                 // q pre-scaled by HK^-0.5
        }
        op[kg][vv] = acc;
        __syncthreads();
        if (tid < 64) {
            float o = op[0][tid] + op[1][tid] + op[2][tid] + op[3][tid];
            op[0][tid] = o; red[tid] = o * o;
        }
        __syncthreads();
        if (tid < 32) red[tid] += red[tid + 32]; __syncthreads();
        if (tid < 16) red[tid] += red[tid + 16]; __syncthreads();
        if (tid < 8)  red[tid] += red[tid + 8];  __syncthreads();
        if (tid < 4)  red[tid] += red[tid + 4];  __syncthreads();
        if (tid < 2)  red[tid] += red[tid + 2];  __syncthreads();
        if (tid < 1)  red[0]   += red[1];        __syncthreads();
        if (tid < 64) {
            float o   = op[0][tid];
            float inv = rsqrtf(red[0] * (1.f / HVz) + 1e-5f);
            float val = o * inv * gnw[tid];
            float gt  = gs[tid];
            val *= gt / (1.f + expf(-gt));          // g * sigmoid(g)
            obf[bv + tid] = (_Float16)val;
        }
        __syncthreads();
    }
}

// ---------------- head: out[b,:] = h[b,-1,:] @ head_w + head_b ----------------
__global__ void head_kernel(const float* __restrict__ h, const float* __restrict__ hw,
                            const float* __restrict__ hb, float* __restrict__ out) {
    int b = blockIdx.x, tid = threadIdx.x;   // 128 threads
    const float* hr = h + ((size_t)b * Tz + (Tz - 1)) * Dz;
    __shared__ float p[3][128];
    float a0 = 0.f, a1 = 0.f, a2 = 0.f;
    for (int d = tid; d < Dz; d += 128) {
        float v = hr[d];
        a0 += v * hw[d * 3 + 0]; a1 += v * hw[d * 3 + 1]; a2 += v * hw[d * 3 + 2];
    }
    p[0][tid] = a0; p[1][tid] = a1; p[2][tid] = a2;
    __syncthreads();
    for (int st = 64; st > 0; st >>= 1) {
        if (tid < st) { p[0][tid] += p[0][tid + st]; p[1][tid] += p[1][tid + st]; p[2][tid] += p[2][tid + st]; }
        __syncthreads();
    }
    if (tid < 3) out[b * 3 + tid] = p[tid][0] + hb[tid];
}

// ---------------- host ----------------
static void launch_gemm(hipStream_t s, const _Float16* A, const _Float16* Wt,
                        const float* bias, const float* resid,
                        float* outF, _Float16* outB,
                        int M, int N, int K, float alpha, int act) {
    if (N % 32 == 0) {
        dim3 grid(M / 32, (N / 32 + 3) / 4);
        gemm_wmma_t<true><<<grid, 128, 0, s>>>(A, Wt, bias, resid, outF, outB, M, N, K, alpha, act);
    } else {
        dim3 grid(M / 32, (N / 16 + 3) / 4);
        gemm_wmma_t<false><<<grid, 128, 0, s>>>(A, Wt, bias, resid, outF, outB, M, N, K, alpha, act);
    }
}

extern "C" void kernel_launch(void* const* d_in, const int* in_sizes, int n_in,
                              void* d_out, int out_size, void* d_ws, size_t ws_size,
                              hipStream_t stream) {
    (void)in_sizes; (void)n_in; (void)out_size; (void)ws_size;
    const float* x     = (const float*)d_in[0];
    const float* bl1   = (const float*)d_in[1];
    const float* bb1   = (const float*)d_in[2];
    const float* bl2   = (const float*)d_in[3];
    const float* bb2   = (const float*)d_in[4];
    const float* by1   = (const float*)d_in[5];
    const float* by2   = (const float*)d_in[6];
    const float* emb_w = (const float*)d_in[7];
    const float* emb_b = (const float*)d_in[8];
    const float* cls   = (const float*)d_in[9];
    const float* ln1w  = (const float*)d_in[10];
    const float* ln1b  = (const float*)d_in[11];
    const float* qw    = (const float*)d_in[12];
    const float* kw    = (const float*)d_in[13];
    const float* vw    = (const float*)d_in[14];
    const float* gw    = (const float*)d_in[15];
    const float* gkw1  = (const float*)d_in[16];
    const float* gkw2  = (const float*)d_in[17];
    const float* gkb2  = (const float*)d_in[18];
    const float* gnw   = (const float*)d_in[19];
    const float* ow    = (const float*)d_in[20];
    const float* ln2w  = (const float*)d_in[21];
    const float* ln2b  = (const float*)d_in[22];
    const float* w1    = (const float*)d_in[23];
    const float* b1m   = (const float*)d_in[24];
    const float* w2    = (const float*)d_in[25];
    const float* b2m   = (const float*)d_in[26];
    const float* hw    = (const float*)d_in[27];
    const float* hb    = (const float*)d_in[28];
    float* out = (float*)d_out;

    char* p = (char*)d_ws;
    auto alloc = [&](size_t bytes) -> char* {
        char* r = p; p += (bytes + 255) & ~(size_t)255; return r;
    };
    const int M = Mz;
    float*    xnorm = (float*)alloc((size_t)Bz * Sz * Fz * 4);
    float*    h     = (float*)alloc((size_t)M * Dz * 4);
    _Float16* xnf   = (_Float16*)alloc((size_t)M * Dz * 2);
    float*    qb    = (float*)alloc((size_t)M * DKz * 4);
    float*    kb    = (float*)alloc((size_t)M * DKz * 4);
    float*    vb    = (float*)alloc((size_t)M * DVz * 4);
    float*    gb    = (float*)alloc((size_t)M * DVz * 4);
    float*    gkb   = (float*)alloc((size_t)M * DKz * 4);
    float*    glow  = (float*)alloc((size_t)M * GLRz * 4);
    _Float16* obf   = (_Float16*)alloc((size_t)M * DVz * 2);
    _Float16* hid   = (_Float16*)alloc((size_t)M * MLPz * 2);
    _Float16 *qwT[Lz], *kwT[Lz], *vwT[Lz], *gwT[Lz], *g1T[Lz], *owT[Lz], *w1T[Lz], *w2T[Lz];
    for (int i = 0; i < Lz; ++i) {
        qwT[i] = (_Float16*)alloc((size_t)Dz * DKz * 2);
        kwT[i] = (_Float16*)alloc((size_t)Dz * DKz * 2);
        vwT[i] = (_Float16*)alloc((size_t)Dz * DVz * 2);
        gwT[i] = (_Float16*)alloc((size_t)Dz * DVz * 2);
        g1T[i] = (_Float16*)alloc((size_t)Dz * GLRz * 2);
        owT[i] = (_Float16*)alloc((size_t)DVz * Dz * 2);
        w1T[i] = (_Float16*)alloc((size_t)Dz * MLPz * 2);
        w2T[i] = (_Float16*)alloc((size_t)MLPz * Dz * 2);
    }

    auto tc = [&](const float* src, _Float16* dst, int K, int N) {
        int total = K * N;
        tcast_f16<<<(total + 255) / 256, 256, 0, stream>>>(src, dst, K, N);
    };
    for (int i = 0; i < Lz; ++i) {
        tc(qw   + (size_t)i * Dz * DKz,   qwT[i], Dz,   DKz);
        tc(kw   + (size_t)i * Dz * DKz,   kwT[i], Dz,   DKz);
        tc(vw   + (size_t)i * Dz * DVz,   vwT[i], Dz,   DVz);
        tc(gw   + (size_t)i * Dz * DVz,   gwT[i], Dz,   DVz);
        tc(gkw1 + (size_t)i * Dz * GLRz,  g1T[i], Dz,   GLRz);
        tc(ow   + (size_t)i * DVz * Dz,   owT[i], DVz,  Dz);
        tc(w1   + (size_t)i * Dz * MLPz,  w1T[i], Dz,   MLPz);
        tc(w2   + (size_t)i * MLPz * Dz,  w2T[i], MLPz, Dz);
    }

    bin_norm_kernel<<<Bz, 256, 0, stream>>>(x, bl1, bb1, bl2, bb2, by1, by2, xnorm);
    embed_kernel<<<dim3(Tz, Bz), 128, 0, stream>>>(xnorm, emb_w, emb_b, cls, h);

    const float qscale = 0.17677669529663687f;   // HK^-0.5 = 1/sqrt(32)
    for (int i = 0; i < Lz; ++i) {
        layernorm_kernel<<<M, 128, 0, stream>>>(h, ln1w + i * Dz, ln1b + i * Dz, xnf);
        launch_gemm(stream, xnf, qwT[i], nullptr, nullptr, qb,   nullptr, M, DKz,  Dz,  qscale, 0);
        launch_gemm(stream, xnf, kwT[i], nullptr, nullptr, kb,   nullptr, M, DKz,  Dz,  1.f,    0);
        launch_gemm(stream, xnf, vwT[i], nullptr, nullptr, vb,   nullptr, M, DVz,  Dz,  1.f,    0);
        launch_gemm(stream, xnf, gwT[i], nullptr, nullptr, gb,   nullptr, M, DVz,  Dz,  1.f,    0);
        launch_gemm(stream, xnf, g1T[i], nullptr, nullptr, glow, nullptr, M, GLRz, Dz,  1.f,    0);
        gk2_kernel<<<M, DKz, 0, stream>>>(glow, gkw2 + (size_t)i * GLRz * DKz, gkb2 + i * DKz, gkb);
        gla_scan_kernel<<<Bz * Hz, 256, 0, stream>>>(qb, kb, vb, gb, gkb, gnw + i * HVz, obf);
        launch_gemm(stream, obf, owT[i], nullptr, h, h, nullptr, M, Dz, DVz, 1.f, 0);
        layernorm_kernel<<<M, 128, 0, stream>>>(h, ln2w + i * Dz, ln2b + i * Dz, xnf);
        launch_gemm(stream, xnf, w1T[i], b1m + (size_t)i * MLPz, nullptr, nullptr, hid, M, MLPz, Dz, 1.f, 1);
        launch_gemm(stream, hid, w2T[i], b2m + i * Dz, h, h, nullptr, M, Dz, MLPz, 1.f, 0);
    }
    head_kernel<<<Bz, 128, 0, stream>>>(h, hw, hb, out);
}
